// MLPGateDGL_18004502904920
// MI455X (gfx1250) — matrix operand order repaired and checked by reading, not compiled, not run on hardware
//
#include <hip/hip_runtime.h>
#include <cstdint>
#include <cstddef>

// ---------------- problem constants ----------------
constexpr int kN  = 262144;
constexpr int kH  = 128;
constexpr int kDM = 256;
constexpr int kL  = 16;
constexpr int kM  = kN / kL;     // 16384
constexpr int kP  = 4096;
constexpr int kTM = 32;          // rows per block tile (2 x 16-row WMMA tiles)
constexpr int kRT = kTM / 16;    // row tiles per block
constexpr int kLDA = 136;        // padded bf16 stride (128 + 8)
constexpr int kLDB = 264;        // padded bf16 stride (256 + 8)

typedef __bf16 bf16;
typedef __attribute__((ext_vector_type(16))) __bf16 v16bf;
typedef __attribute__((ext_vector_type(8)))  float  v8f;
typedef __attribute__((ext_vector_type(4)))  unsigned int v4u;

// ---------------- device helpers ----------------
__device__ __forceinline__ float sigmoidf_(float x) { return 1.f / (1.f + expf(-x)); }

// Load one 16x32 bf16 A-fragment from a row-major LDS tile, matching the CDNA5
// 16-bit A-matrix VGPR layout (lanes 0-15: rows, K 0..7 & 16..23; lanes 16-31: K 8..15 & 24..31).
__device__ __forceinline__ v16bf load_a_frag(const bf16* sA, int lda, int k0) {
  const int lane = threadIdx.x & 31;
  const bf16* p = sA + (lane & 15) * lda + k0 + ((lane >> 4) << 3);
  v16bf a;
  ((v4u*)&a)[0] = *(const v4u*)p;          // ds_load_b128
  ((v4u*)&a)[1] = *(const v4u*)(p + 16);   // ds_load_b128
  return a;
}

// Packed-B fragment pointer: tile*512 bf16, lane*16 within tile.
__device__ __forceinline__ const v16bf* bfrag(const bf16* Bp, int tile) {
  return (const v16bf*)(Bp + (size_t)tile * 512) + (threadIdx.x & 31);
}

// GEMM: out(kTM x Ncols) = A(kTM x K) @ B(K x Ncols) + bias, optional affine +
// ReLU, bf16 output to LDS. Each B fragment is loaded ONCE and reused for all
// RT row-tiles (B-register reuse -> halves L2 weight traffic vs 16-row tiles).
template<int RT, bool RELU, bool GB>
__device__ __forceinline__ void gemm16(const bf16* __restrict__ sA, int lda, int K,
                                       const bf16* __restrict__ Bp,
                                       const float* __restrict__ bias,
                                       const float* __restrict__ gam,
                                       const float* __restrict__ bet,
                                       int Ncols, bf16* __restrict__ outB, int ldo) {
  const int lane = threadIdx.x & 31;
  const int wave = threadIdx.x >> 5;
  const int nT = Ncols >> 4;
  const int kT = K >> 5;
  for (int t = wave; t < nT; t += 8) {
    v8f c[RT];
#pragma unroll
    for (int rt = 0; rt < RT; ++rt) c[rt] = (v8f){};
    const v16bf* bp = bfrag(Bp, t * kT);
    for (int kt = 0; kt < kT; ++kt) {
      v16bf b = *bp;                 // 2x global_load_b128, L2-resident weights
      bp += 32;
#pragma unroll
      for (int rt = 0; rt < RT; ++rt) {
        v16bf a = load_a_frag(sA + rt * 16 * lda, lda, kt << 5);
        c[rt] = __builtin_amdgcn_wmma_f32_16x16x32_bf16(false, a, false, b,
                                                        (short)0, c[rt], false, false);
      }
    }
    const int n = (t << 4) + (lane & 15);
    const float bb = bias[n];
    const float gg = GB ? gam[n] : 0.f;
    const float ee = GB ? bet[n] : 0.f;
    const int mb = (lane >> 4) << 3;
#pragma unroll
    for (int rt = 0; rt < RT; ++rt) {
#pragma unroll
      for (int r = 0; r < 8; ++r) {
        float v = c[rt][r] + bb;
        if (GB)   v = gg * v + ee;
        if (RELU) v = fmaxf(v, 0.f);
        outB[(rt * 16 + mb + r) * ldo + n] = (bf16)v;
      }
    }
  }
}

// Fused GRU stage: wave w computes the ir/iz/inn column-tiles for its 16
// columns (3 accumulators per row-tile), applies the GRU nonlinearity with the
// folded h==1 constants entirely in registers, and writes either the AND
// candidate (LDS) or the gate-selected final rows (global f32).
template<int RT, bool AND>
__device__ __forceinline__ void gru_stage(const bf16* __restrict__ sA, int lda,
                                          const bf16* __restrict__ wih,
                                          const float* __restrict__ bih,
                                          const float* __restrict__ ghc,
                                          const int* __restrict__ gate, int row0,
                                          float* __restrict__ sCand,
                                          float* __restrict__ out) {
  const int lane = threadIdx.x & 31;
  const int wave = threadIdx.x >> 5;            // column group: n0 = wave*16
  const int kT = kH >> 5;                       // K = 128 -> 4 k-tiles
  v8f cr[RT], cz[RT], cn[RT];
#pragma unroll
  for (int rt = 0; rt < RT; ++rt) { cr[rt] = (v8f){}; cz[rt] = (v8f){}; cn[rt] = (v8f){}; }
  const v16bf* bpr = bfrag(wih, (wave)      * kT);   // ir tiles: cols   0..127
  const v16bf* bpz = bfrag(wih, (wave + 8)  * kT);   // iz tiles: cols 128..255
  const v16bf* bpn = bfrag(wih, (wave + 16) * kT);   // inn tiles: cols 256..383
  for (int kt = 0; kt < kT; ++kt) {
    v16bf br = *bpr; bpr += 32;
    v16bf bz = *bpz; bpz += 32;
    v16bf bn = *bpn; bpn += 32;
#pragma unroll
    for (int rt = 0; rt < RT; ++rt) {
      v16bf a = load_a_frag(sA + rt * 16 * lda, lda, kt << 5);
      cr[rt] = __builtin_amdgcn_wmma_f32_16x16x32_bf16(false, a, false, br, (short)0, cr[rt], false, false);
      cz[rt] = __builtin_amdgcn_wmma_f32_16x16x32_bf16(false, a, false, bz, (short)0, cz[rt], false, false);
      cn[rt] = __builtin_amdgcn_wmma_f32_16x16x32_bf16(false, a, false, bn, (short)0, cn[rt], false, false);
    }
  }
  const int n = (wave << 4) + (lane & 15);
  const int mb = (lane >> 4) << 3;
  const float bir = bih[n],       biz = bih[kH + n],  bin = bih[2 * kH + n];
  const float hr  = ghc[n],       hz  = ghc[kH + n],  hn  = ghc[2 * kH + n];
#pragma unroll
  for (int rt = 0; rt < RT; ++rt) {
#pragma unroll
    for (int r = 0; r < 8; ++r) {
      const int m = rt * 16 + mb + r;
      float rg = sigmoidf_(cr[rt][r] + bir + hr);
      float zg = sigmoidf_(cz[rt][r] + biz + hz);
      float ng = tanhf(cn[rt][r] + bin + rg * hn);
      float cand = (1.f - zg) * ng + zg;           // h == 1
      if (AND) {
        sCand[m * kH + n] = cand;
      } else {
        float v = (gate[row0 + m] == 1) ? sCand[m * kH + n] : cand;
        out[(size_t)(row0 + m) * kH + n] = v;
      }
    }
  }
}

// ---------------- weight pre-pack / bias folding ----------------
// dst[tid] layout: tile=tid>>9 (= colTile*kT + kTile), lane=(tid>>4)&31, e=tid&15
// element (k,n): k = kTile*32 + (e>=8)*16 + (lane>=16)*8 + (e&7), n = colTile*16 + (lane&15)
__global__ void pack_b(const float* __restrict__ W, bf16* __restrict__ dst,
                       int K, int Ncols, int kS, int nS) {
  int tid = blockIdx.x * blockDim.x + threadIdx.x;
  if (tid >= K * Ncols) return;
  int tile = tid >> 9, r = tid & 511;
  int l = r >> 4, e = r & 15;
  int kT = K >> 5;
  int ct = tile / kT, kt = tile - ct * kT;
  int k = (kt << 5) + ((e >> 3) << 4) + ((l >> 4) << 3) + (e & 7);
  int n = (ct << 4) + (l & 15);
  dst[tid] = (bf16)W[(size_t)k * kS + (size_t)n * nS];
}

// out[n] = b[n] + sum_k W[k*kS + n*nS]   (folds the all-ones activations)
__global__ void fold_bias(const float* __restrict__ W, const float* __restrict__ b,
                          float* __restrict__ out, int K, int Np, int kS, int nS) {
  int n = blockIdx.x * blockDim.x + threadIdx.x;
  if (n >= Np) return;
  float s = b[n];
  for (int k = 0; k < K; ++k) s += W[(size_t)k * kS + (size_t)n * nS];
  out[n] = s;
}

__global__ void ones_kernel(float* __restrict__ outHs, float* __restrict__ outHf) {
  size_t i = (size_t)blockIdx.x * blockDim.x + threadIdx.x;
  if (i < (size_t)kM * kH) { outHs[i] = 1.f; outHf[i] = 1.f; }
}

// ---------------- fused level kernel ----------------
struct BranchW {
  const bf16 *w1, *w2, *w3, *wih;                    // packed bf16 fragments
  const float *b1, *b2, *b3, *bih, *ghc;             // f32 biases (+folded consts)
};
struct LevelArgs {
  BranchW hsA, hsN, hfA, hfN;
  const float* x;
  const int* gate;
  float* outHs;
  float* outHf;
};

__global__ __launch_bounds__(256) void level_kernel(LevelArgs A) {
  __shared__ bf16  sX[kTM * kLDA];
  __shared__ bf16  sU[kTM * kLDB];
  __shared__ bf16  sV[kTM * kLDB];
  __shared__ float sCand[kTM * kH];

  const int row0 = kM + blockIdx.x * kTM;            // levels 1..15 only
  for (int e = threadIdx.x; e < kTM * kH; e += 256) {
    int m = e >> 7, j = e & 127;
    sX[m * kLDA + j] = (bf16)A.x[(size_t)(row0 + m) * kH + j];
  }
  __syncthreads();

  auto run_mlp = [&](const BranchW& w) {
    // h1 = relu(x @ W1 + b1)          (32x128)@(128x256)
    gemm16<kRT, true, false>(sX, kLDA, kH, w.w1, w.b1, nullptr, nullptr, kDM, sU, kLDB);
    __syncthreads();
    // h2 = relu(h1 @ W2 + b2)         (32x256)@(256x256)
    gemm16<kRT, true, false>(sU, kLDB, kDM, w.w2, w.b2, nullptr, nullptr, kDM, sV, kLDB);
    __syncthreads();
    // m  = h2 @ W3 + b3               (32x256)@(256x128)
    gemm16<kRT, false, false>(sV, kLDB, kDM, w.w3, w.b3, nullptr, nullptr, kH, sU, kLDB);
    __syncthreads();
  };

  // hs: AND branch (candidate to LDS), NOT branch (gate-select to global)
  run_mlp(A.hsA);
  gru_stage<kRT, true >(sU, kLDB, A.hsA.wih, A.hsA.bih, A.hsA.ghc, A.gate, row0, sCand, A.outHs);
  __syncthreads();
  run_mlp(A.hsN);
  gru_stage<kRT, false>(sU, kLDB, A.hsN.wih, A.hsN.bih, A.hsN.ghc, A.gate, row0, sCand, A.outHs);
  __syncthreads();

  // hf: same, with the ones-half of xc folded into b1 and ghc
  run_mlp(A.hfA);
  gru_stage<kRT, true >(sU, kLDB, A.hfA.wih, A.hfA.bih, A.hfA.ghc, A.gate, row0, sCand, A.outHf);
  __syncthreads();
  run_mlp(A.hfN);
  gru_stage<kRT, false>(sU, kLDB, A.hfN.wih, A.hfN.bih, A.hfN.ghc, A.gate, row0, sCand, A.outHf);
}

// ---------------- readout kernels ----------------
struct ROArgs {
  const bf16 *w1, *w2;
  const float *b1, *g1, *e1, *b2, *g2, *e2, *w3, *b3;
};

// final 256 -> 1 projection: 8 segments of 32 per row, LDS reduction
__device__ __forceinline__ void ro_tail(const bf16* sH2, const float* w3,
                                        const float* b3, bool sigm,
                                        float* out, int row0) {
  __shared__ float sRed[kTM][9];
  const int rr = threadIdx.x >> 3, seg = threadIdx.x & 7;
  float p = 0.f;
  for (int i = 0; i < 32; ++i)
    p += (float)sH2[rr * kLDB + seg * 32 + i] * w3[seg * 32 + i];
  sRed[rr][seg] = p;
  __syncthreads();
  if (seg == 0) {
    float s = b3[0];
    for (int i = 0; i < 8; ++i) s += sRed[rr][i];
    out[row0 + rr] = sigm ? 1.f / (1.f + expf(-s)) : s;
  }
}

__global__ __launch_bounds__(256) void ro_prob_kernel(ROArgs R, const float* __restrict__ hf,
                                                      float* __restrict__ prob) {
  __shared__ bf16 sIn[kTM * kLDB];
  __shared__ bf16 sH[kTM * kLDB];
  const int row0 = blockIdx.x * kTM;
  for (int e = threadIdx.x; e < kTM * kH; e += 256) {
    int m = e >> 7, j = e & 127;
    sIn[m * kLDB + j] = (bf16)hf[(size_t)(row0 + m) * kH + j];
  }
  __syncthreads();
  gemm16<kRT, true, true>(sIn, kLDB, kH, R.w1, R.b1, R.g1, R.e1, kDM, sH, kLDB);
  __syncthreads();
  gemm16<kRT, true, true>(sH, kLDB, kDM, R.w2, R.b2, R.g2, R.e2, kDM, sIn, kLDB);
  __syncthreads();
  ro_tail(sIn, R.w3, R.b3, false, prob, row0);
}

__global__ __launch_bounds__(256) void ro_rc_kernel(ROArgs R, const float* __restrict__ hs,
                                                    const long long* __restrict__ idx,
                                                    float* __restrict__ outrc) {
  __shared__ bf16 sIn[kTM * kLDB];
  __shared__ bf16 sH[kTM * kLDB];
  const int p0 = blockIdx.x * kTM;
  for (int e = threadIdx.x; e < kTM * 2 * kH; e += 256) {
    int m = e >> 8, j = e & 255;
    long long src = (j < kH) ? idx[p0 + m] : idx[kP + p0 + m];
    sIn[m * kLDB + j] = (bf16)hs[(size_t)src * kH + (j & 127)];
  }
  __syncthreads();
  gemm16<kRT, true, true>(sIn, kLDB, 2 * kH, R.w1, R.b1, R.g1, R.e1, kDM, sH, kLDB);
  __syncthreads();
  gemm16<kRT, true, true>(sH, kLDB, kDM, R.w2, R.b2, R.g2, R.e2, kDM, sIn, kLDB);
  __syncthreads();
  ro_tail(sIn, R.w3, R.b3, true, outrc, p0);
}

// ---------------- host driver ----------------
extern "C" void kernel_launch(void* const* d_in, const int* in_sizes, int n_in,
                              void* d_out, int out_size, void* d_ws, size_t ws_size,
                              hipStream_t stream) {
  (void)in_sizes; (void)n_in; (void)out_size; (void)ws_size;

  const float*     x     = (const float*)d_in[0];
  const int*       gate  = (const int*)d_in[1];
  const long long* rcidx = (const long long*)d_in[2];

  float* outHs   = (float*)d_out;
  float* outHf   = outHs + (size_t)kN * kH;
  float* outProb = outHf + (size_t)kN * kH;
  float* outRc   = outProb + kN;

  char* ws = (char*)d_ws;
  size_t off = 0;
  auto carveB = [&](size_t elems) -> bf16* {
    bf16* p = (bf16*)(ws + off);
    off += ((elems * sizeof(bf16)) + 255) & ~(size_t)255;
    return p;
  };
  auto carveF = [&](size_t elems) -> float* {
    float* p = (float*)(ws + off);
    off += ((elems * sizeof(float)) + 255) & ~(size_t)255;
    return p;
  };
  auto pack = [&](const void* W, int K, int Ncols, int kS, int nS) -> bf16* {
    bf16* dst = carveB((size_t)K * Ncols);
    int total = K * Ncols;
    pack_b<<<(total + 255) / 256, 256, 0, stream>>>((const float*)W, dst, K, Ncols, kS, nS);
    return dst;
  };

  // mb = MLP tuple base (W1,b1,W2,b2,W3,b3); gb = GRU tuple base (Wih,Whh,bih,bhh)
  auto mkBranch = [&](int mb, int gb, bool func) -> BranchW {
    BranchW w;
    w.w1  = pack(d_in[mb + 0], kH, kDM, kDM, 1);        // func: only first 128 rows of W1
    w.w2  = pack(d_in[mb + 2], kDM, kDM, kDM, 1);
    w.w3  = pack(d_in[mb + 4], kDM, kH, kH, 1);
    w.wih = pack(d_in[gb + 0], kH, 3 * kH, 1, kH);      // B[k][n] = Wih[n*128 + k]
    w.b2  = (const float*)d_in[mb + 3];
    w.b3  = (const float*)d_in[mb + 5];
    w.bih = (const float*)d_in[gb + 2];
    if (func) {   // b1' = b1 + colsum(W1[128:256, :])  (ones half of xc)
      float* c1 = carveF(kDM);
      fold_bias<<<1, 256, 0, stream>>>((const float*)d_in[mb + 0] + (size_t)kH * kDM,
                                       (const float*)d_in[mb + 1], c1, kH, kDM, kDM, 1);
      w.b1 = c1;
    } else {
      w.b1 = (const float*)d_in[mb + 1];
    }
    float* g = carveF(3 * kH);  // ghc = bhh + rowsum(Whh)  (h == ones)
    fold_bias<<<2, 256, 0, stream>>>((const float*)d_in[gb + 1], (const float*)d_in[gb + 3],
                                     g, kH, 3 * kH, 1, kH);
    w.ghc = g;
    return w;
  };

  LevelArgs LA;
  LA.hsA = mkBranch(3, 9, false);
  LA.hfA = mkBranch(13, 19, true);
  LA.hsN = mkBranch(23, 29, false);
  LA.hfN = mkBranch(33, 39, true);
  LA.x = x; LA.gate = gate; LA.outHs = outHs; LA.outHf = outHf;

  ROArgs RP;
  RP.w1 = pack(d_in[43], kH, kDM, kDM, 1);
  RP.w2 = pack(d_in[47], kDM, kDM, kDM, 1);
  RP.b1 = (const float*)d_in[44]; RP.g1 = (const float*)d_in[45]; RP.e1 = (const float*)d_in[46];
  RP.b2 = (const float*)d_in[48]; RP.g2 = (const float*)d_in[49]; RP.e2 = (const float*)d_in[50];
  RP.w3 = (const float*)d_in[51]; RP.b3 = (const float*)d_in[52];

  ROArgs RR;
  RR.w1 = pack(d_in[53], 2 * kH, kDM, kDM, 1);
  RR.w2 = pack(d_in[57], kDM, kDM, kDM, 1);
  RR.b1 = (const float*)d_in[54]; RR.g1 = (const float*)d_in[55]; RR.e1 = (const float*)d_in[56];
  RR.b2 = (const float*)d_in[58]; RR.g2 = (const float*)d_in[59]; RR.e2 = (const float*)d_in[60];
  RR.w3 = (const float*)d_in[61]; RR.b3 = (const float*)d_in[62];

  // level 0 stays at ones
  ones_kernel<<<((size_t)kM * kH + 255) / 256, 256, 0, stream>>>(outHs, outHf);
  // levels 1..15 fused MLP3+GRU (both branches), 32 rows per block
  level_kernel<<<(kN - kM) / kTM, 256, 0, stream>>>(LA);
  // readouts consume hs/hf already written to d_out (same stream => ordered)
  ro_prob_kernel<<<kN / kTM, 256, 0, stream>>>(RP, outHf, outProb);
  ro_rc_kernel<<<kP / kTM, 256, 0, stream>>>(RR, outHs, rcidx, outRc);
}
